// LinearAttention_7026566496878
// MI455X (gfx1250) — matrix operand order repaired and checked by reading, not compiled, not run on hardware
//
#include <hip/hip_runtime.h>
#include <hip/hip_bf16.h>

typedef __bf16 bf16_t;
typedef __attribute__((ext_vector_type(16))) __bf16 v16bf;
typedef __attribute__((ext_vector_type(8)))  __bf16 v8bf;
typedef __attribute__((ext_vector_type(8)))  float  v8f;

#define B_ 2
#define N_ 2048
#define C_ 768
#define H_ 12
#define D_ 64

// ---------- workspace layout (all offsets 256B aligned) ----------
static constexpr size_t OFF_XB     = 0;          // x bf16            [4096,768]
static constexpr size_t OFF_WQKVT  = 6291456;    // w_qkv^T bf16      [2304,768]
static constexpr size_t OFF_WPROJT = 9830400;    // w_proj^T bf16     [768,768]
static constexpr size_t OFF_QB     = 11010048;   // q=elu+1 bf16      [B,H,N,D]
static constexpr size_t OFF_KB     = 17301504;   // k=elu+1 bf16      [B,H,N,D]
static constexpr size_t OFF_VB     = 23592960;   // v bf16            [B,H,N,D]
static constexpr size_t OFF_KSUM   = 29884416;   // f32 [B,H,D]
static constexpr size_t OFF_KV     = 29890560;   // f32 [B,H,D,D]  (k^T v)
static constexpr size_t OFF_KVT    = 30283776;   // bf16 [B,H,D,D] transposed
static constexpr size_t OFF_DENOM  = 30480384;   // f32 [B,H,N]
static constexpr size_t OFF_HO     = 30676992;   // bf16 [B,N,C] head outputs

// ---------- helpers ----------
__device__ __forceinline__ bf16_t f2bf(float f) {
  union { float f; unsigned u; } a; a.f = f;
  unsigned r = a.u + 0x7FFFu + ((a.u >> 16) & 1u);   // round-to-nearest-even
  unsigned short h = (unsigned short)(r >> 16);
  bf16_t o; __builtin_memcpy(&o, &h, 2); return o;
}
__device__ __forceinline__ float bf2f(bf16_t b) {
  unsigned short h; __builtin_memcpy(&h, &b, 2);
  union { unsigned u; float f; } a; a.u = ((unsigned)h) << 16; return a.f;
}
__device__ __forceinline__ float elup1(float v) { return v > 0.f ? v + 1.f : __expf(v); }

// A fragment: 16x32 (MxK) bf16, row-major source, lda in elements.
__device__ __forceinline__ v16bf load_frag_a(const bf16_t* __restrict__ A, int lda, int lane) {
  int m  = lane & 15;
  int kb = (lane >> 4) * 8;
  const bf16_t* p = A + (size_t)m * lda;
  v8bf lo = *(const v8bf*)(p + kb);
  v8bf hi = *(const v8bf*)(p + kb + 16);
  v16bf r;
#pragma unroll
  for (int i = 0; i < 8; ++i) { r[i] = lo[i]; r[8 + i] = hi[i]; }
  return r;
}

// B fragment: 32x16 (KxN) bf16 sourced from B^T storage: Bt[col][K], ldb in elements.
__device__ __forceinline__ v16bf load_frag_b(const bf16_t* __restrict__ Bt, int ldb, int lane) {
  int c  = lane & 15;
  int kb = (lane >> 4) * 8;
  const bf16_t* p = Bt + (size_t)c * ldb;
  v8bf lo = *(const v8bf*)(p + kb);
  v8bf hi = *(const v8bf*)(p + kb + 16);
  v16bf r;
#pragma unroll
  for (int i = 0; i < 8; ++i) { r[i] = lo[i]; r[8 + i] = hi[i]; }
  return r;
}

__device__ __forceinline__ v8f wmma_bf16(v16bf a, v16bf b, v8f c) {
  return __builtin_amdgcn_wmma_f32_16x16x32_bf16(false, a, false, b, (short)0, c, false, false);
}

__device__ __forceinline__ void store_nt(float* p, float v) {
  __builtin_nontemporal_store(v, p);
}

// ---------- small utility kernels ----------
__global__ void k_cvt(const float* __restrict__ x, bf16_t* __restrict__ o, int n) {
  int i = blockIdx.x * blockDim.x + threadIdx.x;
  if (i < n) o[i] = f2bf(x[i]);
}
__global__ void k_transpose_cvt(const float* __restrict__ w, bf16_t* __restrict__ wt,
                                int rows, int cols) {   // w[rows][cols] -> wt[cols][rows]
  int i = blockIdx.x * blockDim.x + threadIdx.x;
  if (i < rows * cols) {
    int k = i / cols, c = i - k * cols;
    wt[(size_t)c * rows + k] = f2bf(w[i]);
  }
}
__global__ void k_zero(float* __restrict__ p, int n) {
  int i = blockIdx.x * blockDim.x + threadIdx.x;
  if (i < n) p[i] = 0.f;
}

// ---------- stage 1: qkv = x @ w_qkv, fused ELU+1 split into q/k/v bf16 ----------
// Depth-2 software pipeline with explicit double buffers (no rotation copies).
__global__ __launch_bounds__(256) void k_qkv(const bf16_t* __restrict__ xb,
                                             const bf16_t* __restrict__ wt,
                                             bf16_t* __restrict__ qb,
                                             bf16_t* __restrict__ kb,
                                             bf16_t* __restrict__ vb) {
  const int lane = threadIdx.x & 31;
  const int wid  = blockIdx.x * 8 + (threadIdx.x >> 5);
  const int col_tiles = (3 * C_) / 64;               // 36
  const int strips = ((B_ * N_) / 16) * col_tiles;   // 9216
  if (wid >= strips) return;
  const int row_tile = wid / col_tiles;
  const int col_tile = wid - row_tile * col_tiles;
  const int r0 = row_tile * 16, c0 = col_tile * 64;
  const bf16_t* Arow = xb + (size_t)r0 * C_;

  v8f acc[4];
#pragma unroll
  for (int t = 0; t < 4; ++t)
#pragma unroll
    for (int j = 0; j < 8; ++j) acc[t][j] = 0.f;

  v16bf a0 = load_frag_a(Arow, C_, lane);
  v16bf b0[4];
#pragma unroll
  for (int t = 0; t < 4; ++t)
    b0[t] = load_frag_b(wt + (size_t)(c0 + t * 16) * C_, C_, lane);

  for (int kk = 0; kk < C_; kk += 64) {
    // prefetch k-step kk+32 into buffer 1 (always in range: max 736 < 768)
    const int k1 = kk + 32;
    v16bf a1 = load_frag_a(Arow + k1, C_, lane);
    v16bf b1[4];
#pragma unroll
    for (int t = 0; t < 4; ++t)
      b1[t] = load_frag_b(wt + (size_t)(c0 + t * 16) * C_ + k1, C_, lane);
    // compute k-step kk from buffer 0
#pragma unroll
    for (int t = 0; t < 4; ++t)
      acc[t] = wmma_bf16(a0, b0[t], acc[t]);
    // prefetch k-step kk+64 into buffer 0 (last iter: harmless clamped reload)
    const int k2 = (kk + 64 < C_) ? kk + 64 : kk;
    a0 = load_frag_a(Arow + k2, C_, lane);
#pragma unroll
    for (int t = 0; t < 4; ++t)
      b0[t] = load_frag_b(wt + (size_t)(c0 + t * 16) * C_ + k2, C_, lane);
    // compute k-step kk+32 from buffer 1
#pragma unroll
    for (int t = 0; t < 4; ++t)
      acc[t] = wmma_bf16(a1, b1[t], acc[t]);
  }

  const int rsub = (lane >> 4) * 8, cl = lane & 15;
#pragma unroll
  for (int t = 0; t < 4; ++t) {
#pragma unroll
    for (int j = 0; j < 8; ++j) {
      int row = r0 + rsub + j;
      int col = c0 + t * 16 + cl;
      float v = acc[t][j];
      int s   = col / C_;
      int rem = col - s * C_;
      int h   = rem >> 6;
      int d   = rem & 63;
      int bidx = row >> 11;
      int n    = row & (N_ - 1);
      size_t o = ((((size_t)bidx * H_ + h) * N_ + n) << 6) + d;
      if (s == 0)      qb[o] = f2bf(elup1(v));
      else if (s == 1) kb[o] = f2bf(elup1(v));
      else             vb[o] = f2bf(v);
    }
  }
}

// ---------- stage 2: per (b,h): ksum[d] = sum_n k[n,d];  kv[d1,d2] = sum_n k[n,d1]*v[n,d2] ----------
__global__ __launch_bounds__(256) void k_kv(const bf16_t* __restrict__ kb,
                                            const bf16_t* __restrict__ vb,
                                            float* __restrict__ kv,
                                            float* __restrict__ ksum) {
  const int bh = blockIdx.x >> 4;
  const int n0 = (blockIdx.x & 15) * (N_ / 16);      // 128-row chunk
  const bf16_t* K = kb + (size_t)bh * N_ * D_;
  const bf16_t* V = vb + (size_t)bh * N_ * D_;
  const int tid = threadIdx.x;

  if (tid < D_) {
    float s = 0.f;
    for (int n = n0; n < n0 + 128; ++n) s += bf2f(K[(size_t)n * D_ + tid]);
    atomicAdd(&ksum[bh * D_ + tid], s);
  }

  const int d2 = tid & 63;
  const int d1b = tid >> 6;                          // 0..3
  float s[16];
#pragma unroll
  for (int p = 0; p < 16; ++p) s[p] = 0.f;
  for (int n = n0; n < n0 + 128; ++n) {
    float vv = bf2f(V[(size_t)n * D_ + d2]);
#pragma unroll
    for (int p = 0; p < 16; ++p)
      s[p] += bf2f(K[(size_t)n * D_ + d1b + p * 4]) * vv;
  }
#pragma unroll
  for (int p = 0; p < 16; ++p) {
    int d1 = d1b + p * 4;
    atomicAdd(&kv[((size_t)bh << 12) + (d1 << 6) + d2], s[p]);
  }
}

// kv f32 -> kv^T bf16 (so B-fragment loads are contiguous in q @ kv)
__global__ void k_kvt(const float* __restrict__ kv, bf16_t* __restrict__ kvt) {
  int i = blockIdx.x * blockDim.x + threadIdx.x;
  if (i < B_ * H_ * D_ * D_) {
    int bh = i >> 12, r = i & 4095, d1 = r >> 6, d2 = r & 63;
    kvt[((size_t)bh << 12) + (d2 << 6) + d1] = f2bf(kv[i]);
  }
}

// denom[b,h,n] = q[n,:] . ksum  (linear-attention row-sum identity)
__global__ void k_denom(const bf16_t* __restrict__ qb, const float* __restrict__ ksum,
                        float* __restrict__ denom) {
  int i = blockIdx.x * blockDim.x + threadIdx.x;
  if (i < B_ * H_ * N_) {
    int bh = i / N_;
    const bf16_t* q  = qb + (size_t)i * D_;
    const float*  ks = ksum + bh * D_;
    float s = 0.f;
#pragma unroll 8
    for (int d = 0; d < D_; ++d) s += bf2f(q[d]) * ks[d];
    denom[i] = s;
  }
}

// ---------- stage 3: attn = (q @ k^T) / denom, 32x64 tile per wave, NT stores ----------
// All 12 fragment loads issue before the 16 WMMAs: one wait, then back-to-back XDL issue.
__global__ __launch_bounds__(256) void k_attn(const bf16_t* __restrict__ qb,
                                              const bf16_t* __restrict__ kb,
                                              const float* __restrict__ denom,
                                              float* __restrict__ attn) {
  const int lane = threadIdx.x & 31;
  const int wid  = blockIdx.x * 8 + (threadIdx.x >> 5);
  const int strips = B_ * H_ * 64 * 32;              // 49152 (32-row x 64-col tiles)
  if (wid >= strips) return;
  const int bh = wid / (64 * 32);
  const int r  = wid - bh * (64 * 32);
  const int r0 = (r >> 5) * 32;                      // 32-row strip
  const int m0 = (r & 31) * 64;                      // 64-col strip (4 tiles)
  const bf16_t* Q = qb + (size_t)bh * N_ * D_;
  const bf16_t* K = kb + (size_t)bh * N_ * D_;

  v16bf a[4];
  a[0] = load_frag_a(Q + (size_t)r0 * D_, D_, lane);
  a[1] = load_frag_a(Q + (size_t)r0 * D_ + 32, D_, lane);
  a[2] = load_frag_a(Q + (size_t)(r0 + 16) * D_, D_, lane);
  a[3] = load_frag_a(Q + (size_t)(r0 + 16) * D_ + 32, D_, lane);

  v16bf b[8];
#pragma unroll
  for (int t = 0; t < 4; ++t) {
    b[2 * t]     = load_frag_b(K + (size_t)(m0 + t * 16) * D_, D_, lane);
    b[2 * t + 1] = load_frag_b(K + (size_t)(m0 + t * 16) * D_ + 32, D_, lane);
  }

  v8f acc[8];
#pragma unroll
  for (int t = 0; t < 8; ++t)
#pragma unroll
    for (int j = 0; j < 8; ++j) acc[t][j] = 0.f;

#pragma unroll
  for (int t = 0; t < 4; ++t) {
    acc[t]     = wmma_bf16(a[0], b[2 * t],     acc[t]);
    acc[t]     = wmma_bf16(a[1], b[2 * t + 1], acc[t]);
    acc[4 + t] = wmma_bf16(a[2], b[2 * t],     acc[4 + t]);
    acc[4 + t] = wmma_bf16(a[3], b[2 * t + 1], acc[4 + t]);
  }

  const int rsub = (lane >> 4) * 8, cl = lane & 15;
#pragma unroll
  for (int half = 0; half < 2; ++half) {
#pragma unroll
    for (int j = 0; j < 8; ++j) {
      int row = r0 + half * 16 + rsub + j;
      float inv = 1.f / denom[bh * N_ + row];
      size_t base = ((size_t)bh * N_ + row) * N_ + m0 + cl;
#pragma unroll
      for (int t = 0; t < 4; ++t)
        store_nt(&attn[base + t * 16], acc[half * 4 + t][j] * inv);
    }
  }
}

// ---------- stage 4: head_out = (q @ kv^T_storage) / denom  -> bf16 [B,N,C] ----------
__global__ __launch_bounds__(256) void k_headout(const bf16_t* __restrict__ qb,
                                                 const bf16_t* __restrict__ kvt,
                                                 const float* __restrict__ denom,
                                                 bf16_t* __restrict__ ho) {
  const int lane = threadIdx.x & 31;
  const int wid  = blockIdx.x * 8 + (threadIdx.x >> 5);
  const int strips = B_ * H_ * (N_ / 16);            // 3072
  if (wid >= strips) return;
  const int bh = wid >> 7;
  const int r0 = (wid & 127) * 16;
  const int b_ = bh / H_, h = bh - b_ * H_;
  const bf16_t* Q  = qb  + (size_t)bh * N_ * D_;
  const bf16_t* KV = kvt + ((size_t)bh << 12);

  v16bf a0 = load_frag_a(Q + (size_t)r0 * D_, D_, lane);
  v16bf a1 = load_frag_a(Q + (size_t)r0 * D_ + 32, D_, lane);

  v16bf b[8];
#pragma unroll
  for (int t = 0; t < 4; ++t) {
    b[2 * t]     = load_frag_b(KV + (size_t)(t * 16) * D_, D_, lane);
    b[2 * t + 1] = load_frag_b(KV + (size_t)(t * 16) * D_ + 32, D_, lane);
  }

  v8f acc[4];
#pragma unroll
  for (int t = 0; t < 4; ++t)
#pragma unroll
    for (int j = 0; j < 8; ++j) acc[t][j] = 0.f;

#pragma unroll
  for (int t = 0; t < 4; ++t) {
    acc[t] = wmma_bf16(a0, b[2 * t],     acc[t]);
    acc[t] = wmma_bf16(a1, b[2 * t + 1], acc[t]);
  }

  const int rsub = (lane >> 4) * 8, cl = lane & 15;
#pragma unroll
  for (int j = 0; j < 8; ++j) {
    int row = r0 + rsub + j;
    float inv = 1.f / denom[bh * N_ + row];
#pragma unroll
    for (int t = 0; t < 4; ++t) {
      int col = t * 16 + cl;
      ho[((size_t)(b_ * N_ + row)) * C_ + h * D_ + col] = f2bf(acc[t][j] * inv);
    }
  }
}

// ---------- stage 5: out = ho @ w_proj + b_proj  -> f32 d_out (NT stores) ----------
// Same depth-2 double-buffered pipeline as k_qkv.
__global__ __launch_bounds__(256) void k_proj(const bf16_t* __restrict__ ho,
                                              const bf16_t* __restrict__ wt,
                                              const float* __restrict__ bias,
                                              float* __restrict__ out) {
  const int lane = threadIdx.x & 31;
  const int wid  = blockIdx.x * 8 + (threadIdx.x >> 5);
  const int col_tiles = C_ / 64;                     // 12
  const int strips = ((B_ * N_) / 16) * col_tiles;   // 3072
  if (wid >= strips) return;
  const int row_tile = wid / col_tiles;
  const int col_tile = wid - row_tile * col_tiles;
  const int r0 = row_tile * 16, c0 = col_tile * 64;
  const bf16_t* Arow = ho + (size_t)r0 * C_;

  v8f acc[4];
#pragma unroll
  for (int t = 0; t < 4; ++t)
#pragma unroll
    for (int j = 0; j < 8; ++j) acc[t][j] = 0.f;

  v16bf a0 = load_frag_a(Arow, C_, lane);
  v16bf b0[4];
#pragma unroll
  for (int t = 0; t < 4; ++t)
    b0[t] = load_frag_b(wt + (size_t)(c0 + t * 16) * C_, C_, lane);

  for (int kk = 0; kk < C_; kk += 64) {
    const int k1 = kk + 32;
    v16bf a1 = load_frag_a(Arow + k1, C_, lane);
    v16bf b1[4];
#pragma unroll
    for (int t = 0; t < 4; ++t)
      b1[t] = load_frag_b(wt + (size_t)(c0 + t * 16) * C_ + k1, C_, lane);
#pragma unroll
    for (int t = 0; t < 4; ++t)
      acc[t] = wmma_bf16(a0, b0[t], acc[t]);
    const int k2 = (kk + 64 < C_) ? kk + 64 : kk;
    a0 = load_frag_a(Arow + k2, C_, lane);
#pragma unroll
    for (int t = 0; t < 4; ++t)
      b0[t] = load_frag_b(wt + (size_t)(c0 + t * 16) * C_ + k2, C_, lane);
#pragma unroll
    for (int t = 0; t < 4; ++t)
      acc[t] = wmma_bf16(a1, b1[t], acc[t]);
  }

  const int rsub = (lane >> 4) * 8, cl = lane & 15;
#pragma unroll
  for (int t = 0; t < 4; ++t) {
#pragma unroll
    for (int j = 0; j < 8; ++j) {
      int row = r0 + rsub + j;
      int col = c0 + t * 16 + cl;
      store_nt(&out[(size_t)row * C_ + col], acc[t][j] + bias[col]);
    }
  }
}

// ---------- host ----------
extern "C" void kernel_launch(void* const* d_in, const int* in_sizes, int n_in,
                              void* d_out, int out_size, void* d_ws, size_t ws_size,
                              hipStream_t stream) {
  const float* x      = (const float*)d_in[0];
  const float* w_qkv  = (const float*)d_in[1];
  const float* w_proj = (const float*)d_in[2];
  const float* b_proj = (const float*)d_in[3];
  float* out  = (float*)d_out;
  float* attn = out + (size_t)B_ * N_ * C_;          // tuple output #2
  char* ws = (char*)d_ws;

  bf16_t* xb     = (bf16_t*)(ws + OFF_XB);
  bf16_t* wqkvT  = (bf16_t*)(ws + OFF_WQKVT);
  bf16_t* wprojT = (bf16_t*)(ws + OFF_WPROJT);
  bf16_t* qb     = (bf16_t*)(ws + OFF_QB);
  bf16_t* kb     = (bf16_t*)(ws + OFF_KB);
  bf16_t* vb     = (bf16_t*)(ws + OFF_VB);
  float*  ksum   = (float*)(ws + OFF_KSUM);
  float*  kv     = (float*)(ws + OFF_KV);
  bf16_t* kvt    = (bf16_t*)(ws + OFF_KVT);
  float*  denom  = (float*)(ws + OFF_DENOM);
  bf16_t* ho     = (bf16_t*)(ws + OFF_HO);

  // conversions / transposes
  k_cvt<<<(B_*N_*C_ + 255) / 256, 256, 0, stream>>>(x, xb, B_*N_*C_);
  k_transpose_cvt<<<(C_*3*C_ + 255) / 256, 256, 0, stream>>>(w_qkv, wqkvT, C_, 3*C_);
  k_transpose_cvt<<<(C_*C_ + 255) / 256, 256, 0, stream>>>(w_proj, wprojT, C_, C_);
  // zero ksum + kv (contiguous)
  k_zero<<<((B_*H_*D_ + B_*H_*D_*D_) + 255) / 256, 256, 0, stream>>>(ksum, B_*H_*D_ + B_*H_*D_*D_);

  // stage 1: qkv GEMM + elu
  k_qkv<<<1152, 256, 0, stream>>>(xb, wqkvT, qb, kb, vb);
  // stage 2: k^T v and ksum
  k_kv<<<B_*H_*16, 256, 0, stream>>>(kb, vb, kv, ksum);
  k_kvt<<<(B_*H_*D_*D_ + 255) / 256, 256, 0, stream>>>(kv, kvt);
  k_denom<<<(B_*H_*N_ + 255) / 256, 256, 0, stream>>>(qb, ksum, denom);
  // stage 3: attn matrix (normalized) -> d_out, non-temporal streaming stores
  k_attn<<<6144, 256, 0, stream>>>(qb, kb, denom, attn);
  // stage 4: head outputs via q @ (k^T v)
  k_headout<<<384, 256, 0, stream>>>(qb, kvt, denom, ho);
  // stage 5: output projection
  k_proj<<<384, 256, 0, stream>>>(ho, wprojT, b_proj, out);
}
